// SelfAttention_13383118095010
// MI455X (gfx1250) — matrix operand order repaired and checked by reading, not compiled, not run on hardware
//
#include <hip/hip_runtime.h>
#include <stdint.h>

// Problem constants
#define Bsz 2
#define Lsz 2048
#define Csz 1024
#define Hsz 16
#define Dsz 64
#define MROWS (Bsz * Lsz)          // 4096
#define N_QKV (3 * Csz)            // 3072
#define LOG100 4.6051702f
#define EPSN 1e-12f

typedef __attribute__((ext_vector_type(16))) _Float16 v16h;
typedef __attribute__((ext_vector_type(8)))  float    v8f;

// ---- cross-lane xor shuffle via ds_swizzle (wave32, group-of-32 mode) ----
template <int MASK>
__device__ __forceinline__ float lane_xor(float v) {
    union { float f; int i; } u;
    u.f = v;
    u.i = __builtin_amdgcn_ds_swizzle(u.i, 0x1f | (MASK << 10));
    return u.f;
}

// ---- CDNA5 async global->LDS copy (16 bytes per lane), ASYNCcnt tracked ----
__device__ __forceinline__ void async_load_b128(uint32_t lds_off, uint64_t gaddr) {
    asm volatile("global_load_async_to_lds_b128 %0, %1, off"
                 :: "v"(lds_off), "v"(gaddr)
                 : "memory");
}
__device__ __forceinline__ void wait_asynccnt0() {
    asm volatile("s_wait_asynccnt 0" ::: "memory");
}

// =====================================================================
// Tiled GEMM: C[M,N] = A[M,K] * W[N,K]^T (+ bias), WMMA f16 -> f32 acc.
// Block tile 64x64, BK=32, 256 threads = 8 waves, 2 wmma tiles per wave.
// BIAS: 0 = none, 1 = qkv concat(q_bias, 0, v_bias), 2 = bias1[n]
// =====================================================================
template <typename Ta, typename To, int BIAS>
__global__ void __launch_bounds__(256)
gemm_wmma_kernel(const Ta* __restrict__ A, const float* __restrict__ W,
                 const float* __restrict__ bias1, const float* __restrict__ bias2,
                 To* __restrict__ C, int M, int N, int K) {
    __shared__ _Float16 As[64][40];
    __shared__ _Float16 Ws[64][40];

    const int tid  = threadIdx.x;
    const int lane = tid & 31;
    const int wave = tid >> 5;
    const int m16  = lane & 15;
    const int hi   = lane >> 4;            // 0 or 1

    const int bm = blockIdx.y * 64;
    const int bn = blockIdx.x * 64;

    const int wm  = wave & 3;              // M tile 0..3
    const int wn0 = (wave >> 2) * 2;       // N tiles wn0, wn0+1

    const int lrow = tid >> 2;             // 0..63
    const int lcol = (tid & 3) * 8;        // 0,8,16,24

    v8f acc[2] = {};

    for (int k0 = 0; k0 < K; k0 += 32) {
        // stage A (convert to f16) and W tiles into LDS
        {
            const Ta* src = A + (size_t)(bm + lrow) * K + k0 + lcol;
            const float* wsrc = W + (size_t)(bn + lrow) * K + k0 + lcol;
            if (k0 + 32 < K) {               // prefetch next K tile
                __builtin_prefetch(src + 32, 0, 1);
                __builtin_prefetch(wsrc + 32, 0, 1);
            }
#pragma unroll
            for (int i = 0; i < 8; ++i) As[lrow][lcol + i] = (_Float16)(float)src[i];
#pragma unroll
            for (int i = 0; i < 8; ++i) Ws[lrow][lcol + i] = (_Float16)wsrc[i];
        }
        __syncthreads();

        // A fragment (16x32): lane<16 -> K 0-15 ; lane>=16 -> K 16-31
        v16h a;
#pragma unroll
        for (int i = 0; i < 8; ++i) {
            a[i]     = As[wm * 16 + m16][hi * 16 + i];
            a[8 + i] = As[wm * 16 + m16][hi * 16 + 8 + i];
        }
        // B fragments (32x16): lane<16 -> col=lane, K 0-15 ; lane>=16 -> K 16-31
        v16h b0, b1;
#pragma unroll
        for (int i = 0; i < 16; ++i) {
            b0[i] = Ws[(wn0    ) * 16 + m16][hi * 16 + i];
            b1[i] = Ws[(wn0 + 1) * 16 + m16][hi * 16 + i];
        }
        acc[0] = __builtin_amdgcn_wmma_f32_16x16x32_f16(false, a, false, b0,
                                                        (short)0, acc[0], false, false);
        acc[1] = __builtin_amdgcn_wmma_f32_16x16x32_f16(false, a, false, b1,
                                                        (short)0, acc[1], false, false);
        __syncthreads();
    }

    // epilogue: C/D layout -> lane<16 rows r, lane>=16 rows 8+r; col = lane&15
#pragma unroll
    for (int j = 0; j < 2; ++j) {
        const int n = bn + (wn0 + j) * 16 + m16;
        float bv = 0.0f;
        if (BIAS == 1) {
            if (n < Csz)            bv = bias1[n];
            else if (n >= 2 * Csz)  bv = bias2[n - 2 * Csz];
        } else if (BIAS == 2) {
            bv = bias1[n];
        }
#pragma unroll
        for (int r = 0; r < 8; ++r) {
            const int m = bm + wm * 16 + r + hi * 8;
            C[(size_t)m * N + n] = (To)(acc[j][r] + bv);
        }
    }
}

// =====================================================================
// L2-normalize q,k (scale q by exp(min(scale_mul, ln100))), split qkv,
// write q,k as [B,H,L,D] f16 and v transposed [B,H,D,L] f16.
// One wave per (b,h,l) row; D=64 -> 2 elements per lane.
// =====================================================================
__global__ void __launch_bounds__(256)
norm_split_kernel(const _Float16* __restrict__ qkv, const float* __restrict__ scale_mul,
                  _Float16* __restrict__ qh, _Float16* __restrict__ kh,
                  _Float16* __restrict__ vt) {
    const int tid  = threadIdx.x;
    const int lane = tid & 31;
    const int wave = tid >> 5;
    const int row  = blockIdx.x * 8 + wave;     // 0 .. B*H*L-1 = bh*L + l
    const int l    = row & (Lsz - 1);
    const int bh   = row >> 11;                 // b*H + h
    const int h    = bh & (Hsz - 1);
    const int b    = bh >> 4;

    const _Float16* base = qkv + ((size_t)(b * Lsz + l)) * N_QKV + h * Dsz;

    const float q0 = (float)base[lane],           q1 = (float)base[lane + 32];
    const float k0 = (float)base[Csz + lane],     k1 = (float)base[Csz + lane + 32];
    const float v0 = (float)base[2 * Csz + lane], v1 = (float)base[2 * Csz + lane + 32];

    float qs = q0 * q0 + q1 * q1;
    float ks = k0 * k0 + k1 * k1;
    qs += lane_xor<1>(qs);  qs += lane_xor<2>(qs);  qs += lane_xor<4>(qs);
    qs += lane_xor<8>(qs);  qs += lane_xor<16>(qs);
    ks += lane_xor<1>(ks);  ks += lane_xor<2>(ks);  ks += lane_xor<4>(ks);
    ks += lane_xor<8>(ks);  ks += lane_xor<16>(ks);

    const float sm = __expf(fminf(scale_mul[h], LOG100));
    const float qn = sm / fmaxf(sqrtf(qs), EPSN);
    const float kn = 1.0f / fmaxf(sqrtf(ks), EPSN);

    const size_t ro = (size_t)row * Dsz;
    qh[ro + lane]      = (_Float16)(q0 * qn);
    qh[ro + lane + 32] = (_Float16)(q1 * qn);
    kh[ro + lane]      = (_Float16)(k0 * kn);
    kh[ro + lane + 32] = (_Float16)(k1 * kn);

    const size_t vo = (size_t)bh * Dsz * Lsz + l;
    vt[vo + (size_t)lane * Lsz]        = (_Float16)v0;
    vt[vo + (size_t)(lane + 32) * Lsz] = (_Float16)v1;
}

// =====================================================================
// Flash attention: one wave per 16-row Q tile; 32 keys per step.
// K/V tiles staged once per block via async global->LDS copies (all 8
// waves share them). S = Q K^T (+bias), online softmax, O += P V.
// Output [B,L,H*D] f16.
// =====================================================================
__global__ void __launch_bounds__(256)
attn_kernel(const _Float16* __restrict__ qh, const _Float16* __restrict__ kh,
            const _Float16* __restrict__ vt, const float* __restrict__ bias,
            _Float16* __restrict__ out) {
    __shared__ __attribute__((aligned(32))) _Float16 Ks[32 * 64];   // [key][d]   4KB
    __shared__ __attribute__((aligned(32))) _Float16 Vts[64 * 32];  // [d][key]   4KB
    __shared__ __attribute__((aligned(32))) _Float16 Ps[8][16][40]; // wave-private P

    const int tid  = threadIdx.x;
    const int lane = tid & 31;
    const int wave = tid >> 5;
    const int m16  = lane & 15;
    const int hi   = lane >> 4;

    const int b  = blockIdx.z;
    const int h  = blockIdx.y;
    const int q0 = blockIdx.x * 128 + wave * 16;

    const size_t bh = (size_t)(b * Hsz + h);
    const _Float16* qb = qh + bh * Lsz * Dsz;
    const _Float16* kb = kh + bh * Lsz * Dsz;
    const _Float16* vb = vt + bh * Dsz * Lsz;

    // LDS byte addresses (low 32 bits of flat shared address == LDS offset)
    const uint32_t ksBase = (uint32_t)(uintptr_t)(&Ks[0]);
    const uint32_t vsBase = (uint32_t)(uintptr_t)(&Vts[0]);
    // per-thread async copy targets: 256 threads x 16B = one 4KB tile each
    const uint32_t ksDst = ksBase + (uint32_t)tid * 16;
    const int      vrow  = tid >> 2;            // 0..63 (d)
    const uint32_t vchk  = (uint32_t)(tid & 3) * 16;
    const uint32_t vsDst = vsBase + (uint32_t)vrow * 64 + vchk;
    const uint64_t kGBase = (uint64_t)(uintptr_t)kb + (uint64_t)tid * 16;
    const uint64_t vGBase = (uint64_t)(uintptr_t)vb + (uint64_t)vrow * (Lsz * 2) + vchk;

    // Q fragments: two K-steps over D=64
    v16h qf0, qf1;
    {
        const _Float16* qr = qb + (size_t)(q0 + m16) * Dsz;
#pragma unroll
        for (int i = 0; i < 8; ++i) {
            qf0[i]     = qr[hi * 16 + i];
            qf0[8 + i] = qr[hi * 16 + 8 + i];
            qf1[i]     = qr[32 + hi * 16 + i];
            qf1[8 + i] = qr[40 + hi * 16 + i];
        }
    }

    v8f o[4] = {};
    float mrow[8], lsum[8];
#pragma unroll
    for (int r = 0; r < 8; ++r) { mrow[r] = -1e30f; lsum[r] = 0.0f; }

    for (int kc = 0; kc < Lsz; kc += 32) {
        // ---- stage K (32x64, contiguous 4KB) and V^T (64x32, strided) tiles ----
        async_load_b128(ksDst, kGBase + (uint64_t)kc * (Dsz * 2));
        async_load_b128(vsDst, vGBase + (uint64_t)kc * 2);
        wait_asynccnt0();
        __syncthreads();

        v8f s[2] = {};
#pragma unroll
        for (int j = 0; j < 2; ++j) {
            const int keyl = j * 16 + m16;
            const v16h bf0 = *(const v16h*)(&Ks[keyl * 64 + hi * 16]);       // d 0..31
            const v16h bf1 = *(const v16h*)(&Ks[keyl * 64 + 32 + hi * 16]);  // d 32..63
            s[j] = __builtin_amdgcn_wmma_f32_16x16x32_f16(false, qf0, false, bf0,
                                                          (short)0, s[j], false, false);
            s[j] = __builtin_amdgcn_wmma_f32_16x16x32_f16(false, qf1, false, bf1,
                                                          (short)0, s[j], false, false);
        }

        // bias add + per-row max within this 32-key chunk
        float mx[8];
#pragma unroll
        for (int r = 0; r < 8; ++r) {
            const int mg = q0 + r + hi * 8;
            s[0][r] += bias[(size_t)mg * Lsz + kc + m16];
            s[1][r] += bias[(size_t)mg * Lsz + kc + 16 + m16];
            mx[r] = fmaxf(s[0][r], s[1][r]);
        }
#pragma unroll
        for (int r = 0; r < 8; ++r) {
            mx[r] = fmaxf(mx[r], lane_xor<1>(mx[r]));
            mx[r] = fmaxf(mx[r], lane_xor<2>(mx[r]));
            mx[r] = fmaxf(mx[r], lane_xor<4>(mx[r]));
            mx[r] = fmaxf(mx[r], lane_xor<8>(mx[r]));
        }

        float alpha[8];
#pragma unroll
        for (int r = 0; r < 8; ++r) {
            const float mnew = fmaxf(mrow[r], mx[r]);
            alpha[r] = __expf(mrow[r] - mnew);
            mrow[r]  = mnew;
        }

        // P = exp(S - m), row sums, stage P tile (C/D layout) into LDS
#pragma unroll
        for (int r = 0; r < 8; ++r) {
            const float p0 = __expf(s[0][r] - mrow[r]);
            const float p1 = __expf(s[1][r] - mrow[r]);
            float rs = p0 + p1;
            rs += lane_xor<1>(rs);
            rs += lane_xor<2>(rs);
            rs += lane_xor<4>(rs);
            rs += lane_xor<8>(rs);
            lsum[r] = lsum[r] * alpha[r] + rs;
            const int rg = r + hi * 8;
            Ps[wave][rg][m16]      = (_Float16)p0;
            Ps[wave][rg][16 + m16] = (_Float16)p1;
        }

        // rescale running output
#pragma unroll
        for (int g = 0; g < 4; ++g)
#pragma unroll
            for (int r = 0; r < 8; ++r) o[g][r] *= alpha[r];

        // re-read P as an A fragment (same-wave LDS is in-order)
        v16h pa;
#pragma unroll
        for (int i = 0; i < 8; ++i) {
            pa[i]     = Ps[wave][m16][hi * 16 + i];
            pa[8 + i] = Ps[wave][m16][hi * 16 + 8 + i];
        }

        // O += P V  (V^T tile in LDS -> contiguous B fragments)
#pragma unroll
        for (int g = 0; g < 4; ++g) {
            const v16h vf = *(const v16h*)(&Vts[(g * 16 + m16) * 32 + hi * 16]);
            o[g] = __builtin_amdgcn_wmma_f32_16x16x32_f16(false, pa, false, vf,
                                                          (short)0, o[g], false, false);
        }
        __syncthreads();   // protect K/V tiles from next iteration's overwrite
    }

    // normalize and write [B, L, H*D] f16
#pragma unroll
    for (int g = 0; g < 4; ++g) {
        const int d = g * 16 + m16;
#pragma unroll
        for (int r = 0; r < 8; ++r) {
            const int mg = q0 + r + hi * 8;
            const float val = o[g][r] / lsum[r];
            out[((size_t)(b * Lsz + mg)) * Csz + h * Dsz + d] = (_Float16)val;
        }
    }
}

// =====================================================================
extern "C" void kernel_launch(void* const* d_in, const int* in_sizes, int n_in,
                              void* d_out, int out_size, void* d_ws, size_t ws_size,
                              hipStream_t stream) {
    const float* x         = (const float*)d_in[0];
    const float* attn_bias = (const float*)d_in[1];
    const float* W_qkv     = (const float*)d_in[2];
    const float* q_bias    = (const float*)d_in[3];
    const float* v_bias    = (const float*)d_in[4];
    const float* scale_mul = (const float*)d_in[5];
    const float* W_proj    = (const float*)d_in[6];
    const float* b_proj    = (const float*)d_in[7];
    float* out = (float*)d_out;

    char* ws = (char*)d_ws;
    _Float16* qkv_h = (_Float16*)(ws);                              // 4096*3072*2 = 24 MB
    _Float16* qh    = (_Float16*)(ws + 24ull * 1024 * 1024);        // 8 MB
    _Float16* kh    = (_Float16*)(ws + 32ull * 1024 * 1024);        // 8 MB
    _Float16* vt    = (_Float16*)(ws + 40ull * 1024 * 1024);        // 8 MB
    _Float16* ah    = (_Float16*)(ws + 48ull * 1024 * 1024);        // 8 MB

    const dim3 blk(256);

    // 1) QKV projection: [4096,1024] x [3072,1024]^T -> f16 [4096,3072]
    gemm_wmma_kernel<float, _Float16, 1>
        <<<dim3(N_QKV / 64, MROWS / 64), blk, 0, stream>>>(
            x, W_qkv, q_bias, v_bias, qkv_h, MROWS, N_QKV, Csz);

    // 2) normalize q,k; split; transpose v
    norm_split_kernel<<<dim3(Bsz * Hsz * Lsz / 8), blk, 0, stream>>>(
        qkv_h, scale_mul, qh, kh, vt);

    // 3) flash attention (async K/V staging into LDS)
    attn_kernel<<<dim3(Lsz / 128, Hsz, Bsz), blk, 0, stream>>>(
        qh, kh, vt, attn_bias, ah);

    // 4) output projection: [4096,1024] x [1024,1024]^T + b_proj -> f32 out
    gemm_wmma_kernel<_Float16, float, 2>
        <<<dim3(Csz / 64, MROWS / 64), blk, 0, stream>>>(
            ah, W_proj, b_proj, nullptr, out, MROWS, Csz, Csz);
}
